// SubPixelConvolution_7026566496318
// MI455X (gfx1250) — matrix operand order repaired and checked by reading, compiled, not run on hardware
//
#include <hip/hip_runtime.h>
#include <hip/hip_bf16.h>

typedef __attribute__((ext_vector_type(16))) __bf16 v16bf;
typedef __attribute__((ext_vector_type(8)))  float  v8f;

__device__ __forceinline__ unsigned short f32_to_bf16(float f) {
    union { float f; unsigned int u; } c; c.f = f;
    unsigned int u = c.u;
    unsigned int r = u + 0x7FFFu + ((u >> 16) & 1u);   // round-to-nearest-even
    return (unsigned short)(r >> 16);
}

// ---------------------------------------------------------------------------
// Pre-kernel: convert W (f32, HWIO [3,3,64,256]) into fragment-ready bf16.
// Fragment f = (kpos*2 + ks)*16 + ntile holds a 32(K)x16(N) B tile in the
// contiguous-K order used by this file (A loads use the same):
//   lanes 0-15 : K = ks*32 + 0..15  (v0={K0,K1} ... v7={K14,K15}), N = lane
//   lanes 16-31: K = ks*32 + 16..31,                              N = lane-16
// Per lane 16 bf16 = 32 contiguous bytes -> two global_load_b128 at use site.
// Total: 9*2*16 fragments * 1024 B = 294912 B in d_ws (L2-resident).
// ---------------------------------------------------------------------------
__global__ __launch_bounds__(256) void convert_weights_kernel(
    const float* __restrict__ W, unsigned short* __restrict__ Wb)
{
    int e = blockIdx.x * 256 + threadIdx.x;
    if (e >= 9 * 2 * 16 * 32 * 16) return;     // 147456 bf16 elements
    int f    = e >> 9;                         // fragment id
    int r    = e & 511;
    int lane = r >> 4;
    int idx  = r & 15;
    int v    = idx >> 1;
    int h    = idx & 1;
    int ntile = f & 15;
    int ks    = (f >> 4) & 1;
    int kpos  = f >> 5;
    int ic = ks * 32 + ((lane >= 16) ? 16 : 0) + v * 2 + h;
    int oc = ntile * 16 + (lane & 15);
    Wb[e] = f32_to_bf16(W[(kpos * 64 + ic) * 256 + oc]);
}

// ---------------------------------------------------------------------------
// Main kernel: one WG = output row-segment (b, i, j0..j0+63), all 256 oc.
// 8 waves; wave w owns N-tiles 2w,2w+1 across 4 M-subtiles of 16 pixels.
// kpos loop rolled (9 iters, 16 wmma each) to keep VGPRs < 256.
// ---------------------------------------------------------------------------
#define XPITCH 72   // padded channel pitch (144 B = 9 x 16B chunks -> no DS conflicts)

__global__ __launch_bounds__(256) void spconv_ps_kernel(
    const float* __restrict__ x, const unsigned short* __restrict__ Wb,
    const float* __restrict__ bias, float* __restrict__ out)
{
    __shared__ unsigned short xT[3 * 66 * XPITCH];  // halo tile, bf16 (28512 B)
    __shared__ float          yT[32 * 256];         // half of result tile (32768 B)

    const int b  = blockIdx.z;
    const int i  = blockIdx.y;
    const int j0 = blockIdx.x * 64;
    const int t  = threadIdx.x;

    // ---- stage input halo (rows i-1..i+1, cols j0-1..j0+64, ch 0..63) ----
    for (int it = t; it < 3 * 66 * 16; it += 256) {
        int kh  = it / (66 * 16);
        int rem = it - kh * (66 * 16);
        int c   = rem >> 4;
        int g   = rem & 15;
        int ri  = i - 1 + kh;
        int cj  = j0 - 1 + c;
        float4 v = make_float4(0.f, 0.f, 0.f, 0.f);
        if (ri >= 0 && ri < 192 && cj >= 0 && cj < 192)
            v = *(const float4*)(x + (((b * 192 + ri) * 192 + cj) * 64 + g * 4));
        unsigned int lo = ((unsigned)f32_to_bf16(v.y) << 16) | f32_to_bf16(v.x);
        unsigned int hi = ((unsigned)f32_to_bf16(v.w) << 16) | f32_to_bf16(v.z);
        *(uint2*)(&xT[(kh * 66 + c) * XPITCH + g * 4]) = make_uint2(lo, hi);
    }
    __syncthreads();

    const int wave = t >> 5;
    const int lane = t & 31;
    const int hi16 = (lane >> 4) & 1;
    const int lmod = lane & 15;
    const int nt0  = wave * 2;

    // Per-lane invariant bases (all loop offsets become compile-time imms)
    const char* aBase = (const char*)xT + (lmod * 144 + hi16 * 32);  // A frag base
    const uint4* wq   = (const uint4*)Wb;
    const int    li   = nt0 * 64 + lane * 2;     // uint4 index for B fragments

    v8f acc[8] = {};                    // acc[ms*2 + n], ms = M-subtile, n = ntile

    #pragma unroll 1
    for (int kpos = 0; kpos < 9; ++kpos) {
        const int kh = kpos / 3, kw = kpos - kh * 3;
        const char*  ap = aBase + (kh * 66 + kw) * 144;       // one VALU add
        const uint4* wk = wq + (size_t)kpos * 2048;           // scalar bump (32 KB)

        #pragma unroll
        for (int ks = 0; ks < 2; ++ks) {
            union { v16bf v; uint4 q[2]; } b0, b1;
            b0.q[0] = wk[li + ks * 1024 + 0];
            b0.q[1] = wk[li + ks * 1024 + 1];
            b1.q[0] = wk[li + ks * 1024 + 64];
            b1.q[1] = wk[li + ks * 1024 + 65];

            #pragma unroll
            for (int ms = 0; ms < 4; ++ms) {
                union { v16bf v; uint4 q[2]; } a;
                a.q[0] = *(const uint4*)(ap + ms * 2304 + ks * 64);
                a.q[1] = *(const uint4*)(ap + ms * 2304 + ks * 64 + 16);
                acc[ms * 2 + 0] = __builtin_amdgcn_wmma_f32_16x16x32_bf16(
                    false, a.v, false, b0.v, (short)0, acc[ms * 2 + 0], false, false);
                acc[ms * 2 + 1] = __builtin_amdgcn_wmma_f32_16x16x32_bf16(
                    false, a.v, false, b1.v, (short)0, acc[ms * 2 + 1], false, false);
            }
        }
    }

    // ---- bias + two-phase pixel-shuffle writeback through 32-row LDS tile ----
    const float bias0 = bias[nt0 * 16 + lmod];
    const float bias1 = bias[nt0 * 16 + 16 + lmod];

    const int k  = t & 63;
    const int jb = t >> 6;                                   // 0..3
    float*       yw = &yT[hi16 * 8 * 256 + nt0 * 16 + lmod]; // C/D row base
    const float* yr = &yT[jb * 256 + k * 4];
    float*       ob = out + (((size_t)b * 384 + 2 * i) * 384 + 2 * (j0 + jb)) * 64 + k;

    #pragma unroll
    for (int phase = 0; phase < 2; ++phase) {
        #pragma unroll
        for (int ms2 = 0; ms2 < 2; ++ms2) {
            const int ms = phase * 2 + ms2;
            #pragma unroll
            for (int r = 0; r < 8; ++r) {
                yw[ms2 * 4096 + r * 256 +  0] = acc[ms * 2 + 0][r] + bias0;
                yw[ms2 * 4096 + r * 256 + 16] = acc[ms * 2 + 1][r] + bias1;
            }
        }
        __syncthreads();

        // out[b, 2i+q, 2(j0+phase*32+jj)+p, k] = yT[jj][k*4 + p*2 + q], jj=jb+4*g2
        #pragma unroll
        for (int s = 0; s < 4; ++s) {
            const int q = s & 1, p = s >> 1;
            #pragma unroll
            for (int g2 = 0; g2 < 8; ++g2) {
                ob[q * 24576 + p * 64 + g2 * 512 + phase * 4096] =
                    yr[g2 * 1024 + p * 2 + q];
            }
        }
        __syncthreads();   // yT reads done before phase 1 overwrites
    }
}

extern "C" void kernel_launch(void* const* d_in, const int* in_sizes, int n_in,
                              void* d_out, int out_size, void* d_ws, size_t ws_size,
                              hipStream_t stream) {
    const float* x    = (const float*)d_in[0];   // [16,192,192,64]
    const float* W    = (const float*)d_in[1];   // [3,3,64,256]
    const float* bias = (const float*)d_in[2];   // [256]
    float* out        = (float*)d_out;           // [16,384,384,64]
    unsigned short* Wb = (unsigned short*)d_ws;  // 294912 B of scratch

    convert_weights_kernel<<<(294912 + 255) / 256, 256, 0, stream>>>(W, Wb);

    dim3 grid(192 / 64, 192, 16);                // (j-tiles, rows, batch) = 9216 WGs
    spconv_ps_kernel<<<grid, 256, 0, stream>>>(x, Wb, bias, out);
}
